// Self_Attentnion_16982300688938
// MI455X (gfx1250) — compile-verified, matrix operand order checked by reading, not run on hardware
//
#include <hip/hip_runtime.h>
#include <hip/hip_bf16.h>

#define HIDDEN 1024
#define SEQ    2048
#define BATCH  4

typedef __attribute__((ext_vector_type(16))) short    v16s;
typedef __attribute__((ext_vector_type(16))) __bf16   v16bf;
typedef __attribute__((ext_vector_type(8)))  float    v8f;
typedef __attribute__((ext_vector_type(8)))  short    v8s;
typedef __attribute__((ext_vector_type(4)))  float    v4f;
typedef __attribute__((ext_vector_type(4)))  short    v4s;
typedef __attribute__((ext_vector_type(4)))  unsigned v4u;
typedef __attribute__((ext_vector_type(8)))  int      v8i;
typedef __attribute__((ext_vector_type(4)))  int      v4i;

// fp32 -> bf16 round-to-nearest-even (bit trick)
__device__ __forceinline__ unsigned short f2bf(float f) {
    unsigned u = __builtin_bit_cast(unsigned, f);
    u += 0x7FFFu + ((u >> 16) & 1u);
    return (unsigned short)(u >> 16);
}

__device__ __forceinline__ v16bf as_bf(v16s s) { return __builtin_bit_cast(v16bf, s); }

// Load one 16-bit WMMA A/B fragment (16x32 / 32x16) for this lane.
// Lanes 0-15 hold K = 0..7 / 16..23; lanes 16-31 hold K = 8..15 / 24..31.
__device__ __forceinline__ v16s ldfrag(const unsigned short* p, int halfOff) {
    v8s lo = *(const v8s*)(p + halfOff);
    v8s hi = *(const v8s*)(p + halfOff + 16);
    v16s f;
#pragma unroll
    for (int i = 0; i < 8; ++i) { f[i] = lo[i]; f[8 + i] = hi[i]; }
    return f;
}

__device__ __forceinline__ v8f wmma_bf16(v16s a, v16s b, v8f c) {
    return __builtin_amdgcn_wmma_f32_16x16x32_bf16(
        false, as_bf(a), false, as_bf(b), (short)0, c, false, false);
}

// Async global -> LDS 16-byte copy (per lane), tracked by ASYNCcnt.
__device__ __forceinline__ void async_cp16(unsigned ldsAddr, unsigned long long gaddr) {
    asm volatile("global_load_async_to_lds_b128 %0, %1, off"
                 :: "v"(ldsAddr), "v"(gaddr) : "memory");
}
__device__ __forceinline__ void wait_async0() {
    asm volatile("s_wait_asynccnt 0" ::: "memory");
}
__device__ __forceinline__ unsigned lds_addr(const void* p) {
    return (unsigned)(uintptr_t)p;
}

// TDM: load a 16-row x 1024-col bf16 tile (32 KB) global -> LDS with one
// tensor_load_to_lds. Descriptor per CDNA5 ISA.
__device__ __forceinline__ void tdm_load_k_tile(unsigned ldsByteAddr,
                                                unsigned long long gaddr) {
    v4u g0;
    g0[0] = 1u;                                               // count=1 (valid)
    g0[1] = ldsByteAddr;                                      // lds_addr
    g0[2] = (unsigned)gaddr;                                  // global_addr[31:0]
    g0[3] = (unsigned)((gaddr >> 32) & 0x01FFFFFFull)         // global_addr[56:32]
            | (2u << 30);                                     // type=2 (image)
    v8i g1;
    g1[0] = (int)(1u << 16);                 // data_size=1 (2 B), no multicast
    g1[1] = (int)(1024u << 16);              // tensor_dim0[15:0] = 1024
    g1[2] = (int)(16u << 16);                // tensor_dim0 hi=0, tensor_dim1 = 16
    g1[3] = (int)(1024u << 16);              // tile_dim0 = 1024
    g1[4] = (int)16;                         // tile_dim1 = 16, tile_dim2 = 0
    g1[5] = (int)1024;                       // tensor_dim0_stride = 1024 elems
    g1[6] = 0;
    g1[7] = 0;
    v4i gz = (v4i){0, 0, 0, 0};
#if defined(__clang_major__) && (__clang_major__ >= 23)
    v8i gz8 = (v8i){0, 0, 0, 0, 0, 0, 0, 0};
    __builtin_amdgcn_tensor_load_to_lds(g0, g1, gz, gz, gz8, 0);
#else
    __builtin_amdgcn_tensor_load_to_lds(g0, g1, gz, gz, 0);
#endif
}

// ---------------- kernel 0: fp32 -> bf16 bulk convert ----------------
__global__ void cvt_f32_bf16(const float* __restrict__ in,
                             unsigned short* __restrict__ out, int n4) {
    int i = blockIdx.x * blockDim.x + threadIdx.x;
    if (i >= n4) return;
    v4f v = *(const v4f*)(in + (size_t)i * 4);
    v4s o;
#pragma unroll
    for (int j = 0; j < 4; ++j) o[j] = (short)f2bf(v[j]);
    *(v4s*)(out + (size_t)i * 4) = o;
}

// ---------------- kernel 1: fused QKV projection ----------------
// 4 waves/block, distinct 16-row X tiles, shared 64-col W block.
// W tile (64 rows x 64 k = 8 KB) double-buffered in LDS via async copies;
// 8 WMMAs per barrier.
__global__ __launch_bounds__(128) void qkv_gemm(
    const unsigned short* __restrict__ Xb,
    const unsigned short* __restrict__ Wq, const unsigned short* __restrict__ Wk,
    const unsigned short* __restrict__ Wv,
    const float* __restrict__ bq, const float* __restrict__ bk,
    const float* __restrict__ bv,
    unsigned short* __restrict__ Q, unsigned short* __restrict__ K,
    unsigned short* __restrict__ Vt)
{
    __shared__ unsigned short Bs[2][64 * 64];   // 2 x 8 KB

    const int tid  = threadIdx.x;
    const int lane = tid & 31;
    const int wave = tid >> 5;
    const int rowBase = (blockIdx.x * 4 + wave) * 16;
    const int colBase = blockIdx.y * 64;
    const int z = blockIdx.z;
    const unsigned short* W = (z == 0) ? Wq : ((z == 1) ? Wk : Wv);
    const float* bias       = (z == 0) ? bq : ((z == 1) ? bk : bv);

    const int l15 = lane & 15;
    const int halfOff = (lane < 16) ? 0 : 8;
    const int hi8 = halfOff;

    // staging: thread t covers W row colBase + t/2, 32 cols from (t&1)*32
    const unsigned long long gStageBase =
        (unsigned long long)(uintptr_t)(W + ((size_t)(colBase + (tid >> 1)) * HIDDEN
                                             + (size_t)(tid & 1) * 32));
    const unsigned shBase = lds_addr(&Bs[0][0]) + (unsigned)tid * 64;

    auto stage = [&](int kChunk, int bufSel) {
        const unsigned dst = shBase + (unsigned)(bufSel * 8192);
        const unsigned long long src = gStageBase + (unsigned long long)kChunk * 2;
#pragma unroll
        for (int i = 0; i < 4; ++i) async_cp16(dst + i * 16, src + i * 16);
    };

    const unsigned short* aRow = Xb + (size_t)(rowBase + l15) * HIDDEN;

    stage(0, 0);
    wait_async0();
    __syncthreads();

    v8f acc[4];
#pragma unroll
    for (int t = 0; t < 4; ++t) acc[t] = (v8f){};

    v16s a0 = ldfrag(aRow, halfOff);
    v16s a1 = ldfrag(aRow + 32, halfOff);
    int buf = 0;
    for (int k = 0; k < HIDDEN; k += 64) {
        const bool more = (k + 64) < HIDDEN;
        if (more) stage(k + 64, buf ^ 1);
        v16s n0 = a0, n1 = a1;
        if (more) {
            n0 = ldfrag(aRow + k + 64, halfOff);
            n1 = ldfrag(aRow + k + 96, halfOff);
        }
        const unsigned short* bBuf = &Bs[buf][0];
        // sub-step 0: load all 4 fragments, then 4 WMMAs
        v16s f[4];
#pragma unroll
        for (int t = 0; t < 4; ++t)
            f[t] = ldfrag(bBuf + (size_t)(t * 16 + l15) * 64, halfOff);
#pragma unroll
        for (int t = 0; t < 4; ++t) acc[t] = wmma_bf16(a0, f[t], acc[t]);
        // sub-step 1
#pragma unroll
        for (int t = 0; t < 4; ++t)
            f[t] = ldfrag(bBuf + (size_t)(t * 16 + l15) * 64 + 32, halfOff);
#pragma unroll
        for (int t = 0; t < 4; ++t) acc[t] = wmma_bf16(a1, f[t], acc[t]);

        a0 = n0; a1 = n1;
        wait_async0();
        __syncthreads();
        buf ^= 1;
    }

    if (z < 2) {
        unsigned short* out = (z == 0) ? Q : K;
#pragma unroll
        for (int t = 0; t < 4; ++t) {
            const int d = colBase + t * 16 + l15;
            const float bv_ = bias[d];
#pragma unroll
            for (int r = 0; r < 8; ++r) {
                const int row = rowBase + r + hi8;
                out[(size_t)row * HIDDEN + d] = f2bf(acc[t][r] + bv_);
            }
        }
    } else {
        const int b = rowBase >> 11;
        const int s = rowBase & 2047;
#pragma unroll
        for (int t = 0; t < 4; ++t) {
            const int d = colBase + t * 16 + l15;
            const float bv_ = bias[d];
            v8s o;
#pragma unroll
            for (int r = 0; r < 8; ++r) o[r] = (short)f2bf(acc[t][r] + bv_);
            *(v8s*)(Vt + ((size_t)b * HIDDEN + d) * SEQ + s + hi8) = o;
        }
    }
}

// ---------------- kernel 2: scores + softmax -> P (bf16) ----------------
// grid (32,1,4): 4 waves/block share streamed K tiles; each 32 KB K tile is
// moved by one TDM tensor_load_to_lds (wave 0), double-buffered, TENSORcnt.
__global__ __launch_bounds__(128) void attn_softmax(
    const unsigned short* __restrict__ Qm,
    const unsigned short* __restrict__ Km,
    unsigned short* __restrict__ Pm)
{
    __shared__ unsigned short Ks[2][16 * HIDDEN];   // 2 x 32 KB

    const int tid  = threadIdx.x;
    const int lane = tid & 31;
    const int wave = tid >> 5;
    const int b = blockIdx.z;
    const int qBase = (blockIdx.x * 4 + wave) * 16;
    const int l15 = lane & 15;
    const int halfOff = (lane < 16) ? 0 : 8;
    const int hi8 = halfOff;

    const unsigned short* kBatch = Km + (size_t)b * SEQ * HIDDEN;
    const unsigned shKbase = lds_addr(&Ks[0][0]);

    // Resident Q fragments for the whole 1024-wide reduction dimension
    const unsigned short* qRow = Qm + (size_t)(b * SEQ + qBase + l15) * HIDDEN;
    v16s qf[32];
#pragma unroll
    for (int d = 0; d < 32; ++d) qf[d] = ldfrag(qRow + d * 32, halfOff);

    const float scale = 0.03125f;   // 1/sqrt(1024)
    float m[8], l[8];
#pragma unroll
    for (int j = 0; j < 8; ++j) { m[j] = -1e30f; l[j] = 0.0f; }

    auto stageK = [&](int kt, int bufSel) {
        if (wave == 0)
            tdm_load_k_tile(shKbase + (unsigned)bufSel * (16 * HIDDEN * 2),
                            (unsigned long long)(uintptr_t)(kBatch + (size_t)kt * 16 * HIDDEN));
    };
    auto flip = [&]() {
        if (wave == 0) __builtin_amdgcn_s_wait_tensorcnt(0);
        __syncthreads();
    };
    auto score = [&](const unsigned short* kLds) -> v8f {
        v8f c0 = (v8f){}, c1 = (v8f){};
#pragma unroll
        for (int d = 0; d < 32; d += 2) {
            v16s f0 = ldfrag(kLds + (size_t)d * 32, halfOff);
            v16s f1 = ldfrag(kLds + (size_t)(d + 1) * 32, halfOff);
            c0 = wmma_bf16(qf[d], f0, c0);
            c1 = wmma_bf16(qf[d + 1], f1, c1);
        }
        return c0 + c1;
    };

    // ---- pass 1: online row max + row sum ----
    stageK(0, 0);
    flip();
    int buf = 0;
    for (int kt = 0; kt < SEQ / 16; ++kt) {
        if (kt + 1 < SEQ / 16) stageK(kt + 1, buf ^ 1);
        v8f c = score(&Ks[buf][(size_t)l15 * HIDDEN]);
#pragma unroll
        for (int j = 0; j < 8; ++j) {
            float s = c[j] * scale;
            float t = s;
            t = fmaxf(t, __shfl_xor(t, 1));
            t = fmaxf(t, __shfl_xor(t, 2));
            t = fmaxf(t, __shfl_xor(t, 4));
            t = fmaxf(t, __shfl_xor(t, 8));
            const float mn = fmaxf(m[j], t);
            float p = __expf(s - mn);
            p += __shfl_xor(p, 1);
            p += __shfl_xor(p, 2);
            p += __shfl_xor(p, 4);
            p += __shfl_xor(p, 8);
            l[j] = l[j] * __expf(m[j] - mn) + p;
            m[j] = mn;
        }
        flip();
        buf ^= 1;
    }

    float inv[8];
#pragma unroll
    for (int j = 0; j < 8; ++j) inv[j] = 1.0f / l[j];

    // ---- pass 2: recompute scores, write normalized probabilities ----
    stageK(0, 0);
    flip();
    buf = 0;
    for (int kt = 0; kt < SEQ / 16; ++kt) {
        if (kt + 1 < SEQ / 16) stageK(kt + 1, buf ^ 1);
        v8f c = score(&Ks[buf][(size_t)l15 * HIDDEN]);
#pragma unroll
        for (int j = 0; j < 8; ++j) {
            const float p = __expf(c[j] * scale - m[j]) * inv[j];
            const int row = qBase + j + hi8;
            Pm[((size_t)b * SEQ + row) * SEQ + kt * 16 + l15] = f2bf(p);
        }
        flip();
        buf ^= 1;
    }
}

// ---------------- kernel 3: O = P @ V (fp32 out) ----------------
// grid (32,16,4): waves share the Vt tile (64 rows x 64 k = 8 KB),
// double-buffered async staging; 8 WMMAs per barrier.
__global__ __launch_bounds__(128) void pv_gemm(
    const unsigned short* __restrict__ Pm,
    const unsigned short* __restrict__ Vt,
    float* __restrict__ O)
{
    __shared__ unsigned short Vs[2][64 * 64];   // 2 x 8 KB

    const int tid  = threadIdx.x;
    const int lane = tid & 31;
    const int wave = tid >> 5;
    const int b = blockIdx.z;
    const int qBase = (blockIdx.x * 4 + wave) * 16;
    const int colBase = blockIdx.y * 64;
    const int l15 = lane & 15;
    const int halfOff = (lane < 16) ? 0 : 8;
    const int hi8 = halfOff;

    const unsigned short* aRow = Pm + (size_t)(b * SEQ + qBase + l15) * SEQ;
    const unsigned long long gStageBase =
        (unsigned long long)(uintptr_t)(Vt + ((size_t)b * HIDDEN + colBase + (tid >> 1)) * SEQ
                                        + (size_t)(tid & 1) * 32);
    const unsigned shBase = lds_addr(&Vs[0][0]) + (unsigned)tid * 64;

    auto stage = [&](int kChunk, int bufSel) {
        const unsigned dst = shBase + (unsigned)(bufSel * 8192);
        const unsigned long long src = gStageBase + (unsigned long long)kChunk * 2;
#pragma unroll
        for (int i = 0; i < 4; ++i) async_cp16(dst + i * 16, src + i * 16);
    };

    stage(0, 0);
    wait_async0();
    __syncthreads();

    v8f acc[4];
#pragma unroll
    for (int t = 0; t < 4; ++t) acc[t] = (v8f){};

    v16s a0 = ldfrag(aRow, halfOff);
    v16s a1 = ldfrag(aRow + 32, halfOff);
    int buf = 0;
    for (int k = 0; k < SEQ; k += 64) {
        const bool more = (k + 64) < SEQ;
        if (more) stage(k + 64, buf ^ 1);
        v16s n0 = a0, n1 = a1;
        if (more) {
            n0 = ldfrag(aRow + k + 64, halfOff);
            n1 = ldfrag(aRow + k + 96, halfOff);
        }
        const unsigned short* bBuf = &Vs[buf][0];
        v16s f[4];
#pragma unroll
        for (int t = 0; t < 4; ++t)
            f[t] = ldfrag(bBuf + (size_t)(t * 16 + l15) * 64, halfOff);
#pragma unroll
        for (int t = 0; t < 4; ++t) acc[t] = wmma_bf16(a0, f[t], acc[t]);
#pragma unroll
        for (int t = 0; t < 4; ++t)
            f[t] = ldfrag(bBuf + (size_t)(t * 16 + l15) * 64 + 32, halfOff);
#pragma unroll
        for (int t = 0; t < 4; ++t) acc[t] = wmma_bf16(a1, f[t], acc[t]);

        a0 = n0; a1 = n1;
        wait_async0();
        __syncthreads();
        buf ^= 1;
    }

#pragma unroll
    for (int t = 0; t < 4; ++t) {
        const int d = colBase + t * 16 + l15;
#pragma unroll
        for (int r = 0; r < 8; ++r) {
            const int row = b * SEQ + qBase + r + hi8;
            O[(size_t)row * HIDDEN + d] = acc[t][r];
        }
    }
}

// ---------------- host-side launcher ----------------
extern "C" void kernel_launch(void* const* d_in, const int* in_sizes, int n_in,
                              void* d_out, int out_size, void* d_ws, size_t ws_size,
                              hipStream_t stream) {
    const float* X  = (const float*)d_in[0];
    const float* Wq = (const float*)d_in[1];
    const float* bq = (const float*)d_in[2];
    const float* Wk = (const float*)d_in[3];
    const float* bk = (const float*)d_in[4];
    const float* Wv = (const float*)d_in[5];
    const float* bv = (const float*)d_in[6];
    float* O = (float*)d_out;

    const size_t nX = (size_t)BATCH * SEQ * HIDDEN;   // 8388608
    const size_t nW = (size_t)HIDDEN * HIDDEN;        // 1048576
    unsigned short* Xb  = (unsigned short*)d_ws;
    unsigned short* Wqb = Xb  + nX;
    unsigned short* Wkb = Wqb + nW;
    unsigned short* Wvb = Wkb + nW;
    unsigned short* Qw  = Wvb + nW;
    unsigned short* Kw  = Qw  + nX;
    unsigned short* Vtw = Kw  + nX;
    unsigned short* Pw  = Vtw + nX;

    {
        int n4 = (int)(nX / 4);
        cvt_f32_bf16<<<(n4 + 255) / 256, 256, 0, stream>>>(X, Xb, n4);
        int w4 = (int)(nW / 4);
        cvt_f32_bf16<<<(w4 + 255) / 256, 256, 0, stream>>>(Wq, Wqb, w4);
        cvt_f32_bf16<<<(w4 + 255) / 256, 256, 0, stream>>>(Wk, Wkb, w4);
        cvt_f32_bf16<<<(w4 + 255) / 256, 256, 0, stream>>>(Wv, Wvb, w4);
    }

    qkv_gemm<<<dim3(128, 16, 3), 128, 0, stream>>>(
        Xb, Wqb, Wkb, Wvb, bq, bk, bv, Qw, Kw, Vtw);

    attn_softmax<<<dim3(32, 1, 4), 128, 0, stream>>>(Qw, Kw, Pw);

    pv_gemm<<<dim3(32, 16, 4), 128, 0, stream>>>(Pw, Vtw, O);
}
